// ComplexityAttention_90383291777465
// MI455X (gfx1250) — compile-verified, matrix operand order checked by reading, not run on hardware
//
#include <hip/hip_runtime.h>
#include <stddef.h>

#define S_LEN 2048
#define HIDN  2048
#define NH    16
#define NKV   4
#define HD    128
#define GQA   (NH / NKV)

typedef __attribute__((ext_vector_type(16))) __bf16 bf16x16;
typedef __attribute__((ext_vector_type(8)))  __bf16 bf16x8;
typedef __attribute__((ext_vector_type(8)))  float  f32x8;

#define NEG_INF (-__builtin_inff())

__device__ __forceinline__ unsigned short f32_to_bf16(float f) {
    union { float f; unsigned u; } v; v.f = f;
    unsigned r = v.u + 0x7FFFu + ((v.u >> 16) & 1u);  // round-to-nearest-even
    return (unsigned short)(r >> 16);
}

__device__ __forceinline__ f32x8 zero_f32x8() {
    f32x8 v;
#pragma unroll
    for (int i = 0; i < 8; ++i) v[i] = 0.0f;
    return v;
}

// Build one lane's 16x32 bf16 WMMA operand fragment from an LDS row.
// Lane l (<16) holds row M=l with K={0..7,16..23}; lane l+16 holds K={8..15,24..31}.
__device__ __forceinline__ bf16x16 frag_ld(const unsigned short* row_base, int half) {
    bf16x8 lo = *(const bf16x8*)(row_base + half * 8);
    bf16x8 hi = *(const bf16x8*)(row_base + 16 + half * 8);
    return __builtin_shufflevector(lo, hi, 0,1,2,3,4,5,6,7,8,9,10,11,12,13,14,15);
}

// ---------------- fp32 -> bf16 conversion ----------------
__global__ __launch_bounds__(256) void cvt_f32_to_bf16(const float* __restrict__ src,
                                                       unsigned short* __restrict__ dst,
                                                       int n) {
    int i = blockIdx.x * 256 + threadIdx.x;
    if (i < n) dst[i] = f32_to_bf16(src[i]);
}

// ---------------- NT bf16 GEMM: C[m,n] = sum_k A[m,k]*B[n,k] (+ optional 2nd pair) ---
// Double-buffered LDS software pipeline: loads for step k+1 overlap WMMAs of step k.
#define BM  128
#define BN  128
#define BKK 32
#define LDT 40   // 32 + 8 pad (80 bytes/row, 16B-aligned)

__global__ __launch_bounds__(256) void gemm_nt_bf16(
    const unsigned short* __restrict__ A0, const unsigned short* __restrict__ B0,
    const unsigned short* __restrict__ A1, const unsigned short* __restrict__ B1,
    float* __restrict__ C, int N, int K)
{
    __shared__ unsigned short As[2][BM * LDT];
    __shared__ unsigned short Bs[2][BM * LDT];

    const int tid  = threadIdx.x;
    const int lane = tid & 31;
    const int wid  = tid >> 5;       // 8 waves
    const int wm   = wid >> 1;       // 0..3 -> 32-row slab
    const int wn   = wid & 1;        // 0..1 -> 64-col slab
    const int l15  = lane & 15;
    const int half = lane >> 4;

    const int block_row = blockIdx.y * BM;
    const int block_col = blockIdx.x * BN;

    const int lrow = tid >> 1;          // 128 rows, 2 threads per row
    const int lcol = (tid & 1) * 16;    // 16 bf16 each

    f32x8 acc[2][4];
#pragma unroll
    for (int i = 0; i < 2; ++i)
#pragma unroll
        for (int j = 0; j < 4; ++j) acc[i][j] = zero_f32x8();

    const int npair = (A1 != nullptr) ? 2 : 1;
    const int nkA = K / BKK;
    const int nk  = npair * nkA;

    bf16x8 ra0, ra1, rb0, rb1;

    auto gload = [&](int kk) {
        const int pair = (kk >= nkA) ? 1 : 0;
        const int k0   = (kk - pair * nkA) * BKK;
        const unsigned short* Ag = pair ? A1 : A0;
        const unsigned short* Bg = pair ? B1 : B0;
        const unsigned short* asrc = Ag + (size_t)(block_row + lrow) * K + k0 + lcol;
        const unsigned short* bsrc = Bg + (size_t)(block_col + lrow) * K + k0 + lcol;
        if (k0 + BKK < K) {
            __builtin_prefetch(asrc + BKK, 0, 1);
            __builtin_prefetch(bsrc + BKK, 0, 1);
        }
        ra0 = *(const bf16x8*)(asrc);
        ra1 = *(const bf16x8*)(asrc + 8);
        rb0 = *(const bf16x8*)(bsrc);
        rb1 = *(const bf16x8*)(bsrc + 8);
    };
    auto sstore = [&](int buf) {
        *(bf16x8*)&As[buf][lrow * LDT + lcol]     = ra0;
        *(bf16x8*)&As[buf][lrow * LDT + lcol + 8] = ra1;
        *(bf16x8*)&Bs[buf][lrow * LDT + lcol]     = rb0;
        *(bf16x8*)&Bs[buf][lrow * LDT + lcol + 8] = rb1;
    };

    gload(0);
    sstore(0);

    for (int kk = 0; kk < nk; ++kk) {
        const int cur = kk & 1;
        __syncthreads();
        if (kk + 1 < nk) gload(kk + 1);   // overlap next tile's HBM latency with WMMAs

        bf16x16 af[2], bfv[4];
#pragma unroll
        for (int tm = 0; tm < 2; ++tm)
            af[tm] = frag_ld(&As[cur][(wm * 32 + tm * 16 + l15) * LDT], half);
#pragma unroll
        for (int tn = 0; tn < 4; ++tn)
            bfv[tn] = frag_ld(&Bs[cur][(wn * 64 + tn * 16 + l15) * LDT], half);
#pragma unroll
        for (int tm = 0; tm < 2; ++tm)
#pragma unroll
            for (int tn = 0; tn < 4; ++tn)
                acc[tm][tn] = __builtin_amdgcn_wmma_f32_16x16x32_bf16(
                    false, af[tm], false, bfv[tn], (short)0, acc[tm][tn], false, false);

        if (kk + 1 < nk) sstore((kk + 1) & 1);
    }

    // C/D layout: VGPR r -> M = r + 8*half, N = lane&15
#pragma unroll
    for (int tm = 0; tm < 2; ++tm)
#pragma unroll
        for (int tn = 0; tn < 4; ++tn) {
            const int row0 = block_row + wm * 32 + tm * 16 + half * 8;
            const int col  = block_col + wn * 64 + tn * 16 + l15;
#pragma unroll
            for (int r = 0; r < 8; ++r)
                C[(size_t)(row0 + r) * N + col] = acc[tm][tn][r];
        }
}

// ---------------- per-head RMSNorm + RoPE + relayout [S,H*D] f32 -> [H,S,D] bf16 ----
__global__ __launch_bounds__(128) void norm_rope_relayout(
    const float* __restrict__ src, const float* __restrict__ w,
    unsigned short* __restrict__ dst, int H, int do_rope)
{
    const int s = blockIdx.x;
    const int h = blockIdx.y;
    const int d = threadIdx.x;   // 0..127

    __shared__ float red[128];
    __shared__ float yv[128];

    float x = src[((size_t)s * H + h) * HD + d];
    float y = x;
    if (w != nullptr) {
        red[d] = x * x;
        __syncthreads();
#pragma unroll
        for (int off = 64; off > 0; off >>= 1) {
            if (d < off) red[d] += red[d + off];
            __syncthreads();
        }
        y = x * rsqrtf(red[0] * (1.0f / 128.0f) + 1e-6f) * w[d];
    }
    yv[d] = y;
    __syncthreads();
    if (do_rope) {
        const int dh = d & 63;
        const float inv = expf(-(float)dh * 0.14391156831212787f);  // ln(1e4)/64
        const float ang = (float)s * inv;
        const float rot = (d < 64) ? -yv[d + 64] : yv[d - 64];
        y = y * cosf(ang) + rot * sinf(ang);
    }
    dst[((size_t)h * S_LEN + s) * HD + d] = f32_to_bf16(y);
}

// ---------------- causal GQA flash attention (bf16 WMMA, f32 accum) -----------------
// Double-buffered K/V tiles: next tile's loads overlap QK^T + softmax + PV.
#define BR  64
#define BC  32
#define QLD 136
#define KLD 136
#define VLD 40
#define PLD 40

__global__ __launch_bounds__(128) void flash_attn_bf16(
    const unsigned short* __restrict__ Qg,   // [NH, S, HD]
    const unsigned short* __restrict__ Kg,   // [NKV, S, HD]
    const unsigned short* __restrict__ Vg,   // [NKV, S, HD]
    unsigned short* __restrict__ Og)         // [S, NH*HD]
{
    __shared__ unsigned short Qs[BR * QLD];           // 17.0 KB
    __shared__ unsigned short Ks[2][BC * KLD];        // 17.0 KB
    __shared__ unsigned short Vts[2][HD * VLD];       // 20.0 KB, V transposed: [d][n]
    __shared__ unsigned short Ps[4 * 16 * PLD];       //  5.0 KB per-wave P scratch

    const int tid  = threadIdx.x;    // 128 = 4 waves
    const int lane = tid & 31;
    const int wid  = tid >> 5;       // 0..3, 16 q-rows each
    const int l15  = lane & 15;
    const int half = lane >> 4;

    const int i0 = blockIdx.x * BR;
    const int h  = blockIdx.y;
    const int kv = h / GQA;

    // cooperative Q tile load [BR x HD]
    {
        const int r  = tid >> 1;
        const int c0 = (tid & 1) * 64;
        const unsigned short* src = Qg + ((size_t)h * S_LEN + (i0 + r)) * HD + c0;
#pragma unroll
        for (int c = 0; c < 64; c += 8)
            *(bf16x8*)&Qs[r * QLD + c0 + c] = *(const bf16x8*)(src + c);
    }

    // K/V tile loaders (pipelined)
    const int n  = tid >> 2;          // 0..31: KV row within tile
    const int c0 = (tid & 3) * 32;    // 32 elems of d per thread
    bf16x8 kreg[4], vreg[4];
    auto kv_gload = [&](int j) {
        const unsigned short* ksrc = Kg + ((size_t)kv * S_LEN + j * BC + n) * HD + c0;
        const unsigned short* vsrc = Vg + ((size_t)kv * S_LEN + j * BC + n) * HD + c0;
#pragma unroll
        for (int c = 0; c < 4; ++c) kreg[c] = *(const bf16x8*)(ksrc + 8 * c);
#pragma unroll
        for (int c = 0; c < 4; ++c) vreg[c] = *(const bf16x8*)(vsrc + 8 * c);
    };
    auto kv_sstore = [&](int buf) {
#pragma unroll
        for (int c = 0; c < 4; ++c)
            *(bf16x8*)&Ks[buf][n * KLD + c0 + 8 * c] = kreg[c];
        __bf16* vt = (__bf16*)Vts[buf];
#pragma unroll
        for (int c = 0; c < 32; ++c)                 // transpose on store
            vt[(c0 + c) * VLD + n] = vreg[c >> 3][c & 7];
    };

    const int jend = i0 / BC + 2;   // last KV block touching row i0+BR-1 (causal)
    kv_gload(0);
    kv_sstore(0);
    __syncthreads();                // Q tile + K/V tile 0 visible

    // per-wave Q fragments (16 rows x 128, 4 K-chunks)
    bf16x16 qf[4];
#pragma unroll
    for (int kc = 0; kc < 4; ++kc)
        qf[kc] = frag_ld(&Qs[(wid * 16 + l15) * QLD + kc * 32], half);

    f32x8 oacc[8];
#pragma unroll
    for (int t = 0; t < 8; ++t) oacc[t] = zero_f32x8();

    float m_run[8], l_run[8];
#pragma unroll
    for (int r = 0; r < 8; ++r) { m_run[r] = NEG_INF; l_run[r] = 0.0f; }

    unsigned short* Pw = &Ps[wid * 16 * PLD];
    const float scale = 0.08838834764831845f;   // 1/sqrt(128)

    for (int j = 0; j < jend; ++j) {
        const int cur = j & 1;
        __syncthreads();
        if (j + 1 < jend) kv_gload(j + 1);   // overlap with QK^T/softmax/PV below

        // S = Q K^T : this wave's 16 rows x 32 cols
        f32x8 sacc[2];
        sacc[0] = zero_f32x8();
        sacc[1] = zero_f32x8();
#pragma unroll
        for (int kc = 0; kc < 4; ++kc)
#pragma unroll
            for (int tn = 0; tn < 2; ++tn) {
                bf16x16 kfrag = frag_ld(&Ks[cur][(tn * 16 + l15) * KLD + kc * 32], half);
                sacc[tn] = __builtin_amdgcn_wmma_f32_16x16x32_bf16(
                    false, qf[kc], false, kfrag, (short)0, sacc[tn], false, false);
            }

        // online softmax; P -> per-wave LDS as bf16
#pragma unroll
        for (int r = 0; r < 8; ++r) {
            const int rowg = i0 + wid * 16 + r + 8 * half;
            const int cg   = j * BC + l15;
            float s0 = (cg      <= rowg) ? sacc[0][r] * scale : NEG_INF;
            float s1 = (cg + 16 <= rowg) ? sacc[1][r] * scale : NEG_INF;
            float mx = fmaxf(s0, s1);
#pragma unroll
            for (int off = 1; off < 16; off <<= 1)
                mx = fmaxf(mx, __shfl_xor(mx, off, 32));
            const float mn = fmaxf(m_run[r], mx);
            float corr, p0, p1;
            if (mn == NEG_INF) { corr = 1.0f; p0 = 0.0f; p1 = 0.0f; }
            else {
                corr = expf(m_run[r] - mn);
                p0 = (s0 == NEG_INF) ? 0.0f : expf(s0 - mn);
                p1 = (s1 == NEG_INF) ? 0.0f : expf(s1 - mn);
            }
            float ps = p0 + p1;
#pragma unroll
            for (int off = 1; off < 16; off <<= 1)
                ps += __shfl_xor(ps, off, 32);
            l_run[r] = l_run[r] * corr + ps;
            m_run[r] = mn;
#pragma unroll
            for (int tn = 0; tn < 8; ++tn) oacc[tn][r] *= corr;
            Pw[(r + 8 * half) * PLD + l15]      = f32_to_bf16(p0);
            Pw[(r + 8 * half) * PLD + 16 + l15] = f32_to_bf16(p1);
        }

        // same-wave LDS RAW: wait for P stores before re-reading as WMMA operand
        asm volatile("s_wait_dscnt 0" ::: "memory");

        // O += P * V  (A = P [16x32], B = Vt rows [HD x BC])
        bf16x16 pf = frag_ld(&Pw[l15 * PLD], half);
#pragma unroll
        for (int tn = 0; tn < 8; ++tn) {
            bf16x16 vfrag = frag_ld(&Vts[cur][(tn * 16 + l15) * VLD], half);
            oacc[tn] = __builtin_amdgcn_wmma_f32_16x16x32_bf16(
                false, pf, false, vfrag, (short)0, oacc[tn], false, false);
        }

        if (j + 1 < jend) kv_sstore((j + 1) & 1);
    }

    // epilogue: O / l -> bf16, [S, NH*HD] layout for the wo projection
#pragma unroll
    for (int r = 0; r < 8; ++r) {
        const float rl = 1.0f / l_run[r];
        const int sg = i0 + wid * 16 + r + 8 * half;
#pragma unroll
        for (int tn = 0; tn < 8; ++tn) {
            const int dg = tn * 16 + l15;
            Og[(size_t)sg * (NH * HD) + h * HD + dg] = f32_to_bf16(oacc[tn][r] * rl);
        }
    }
}

// ---------------- host-side orchestration ----------------
extern "C" void kernel_launch(void* const* d_in, const int* in_sizes, int n_in,
                              void* d_out, int out_size, void* d_ws, size_t ws_size,
                              hipStream_t stream)
{
    (void)in_sizes; (void)n_in; (void)out_size; (void)ws_size;
    const float* x   = (const float*)d_in[0];
    const float* mu  = (const float*)d_in[1];
    const float* wq  = (const float*)d_in[2];
    const float* wk  = (const float*)d_in[3];
    const float* wv  = (const float*)d_in[4];
    const float* wo  = (const float*)d_in[5];
    const float* wmq = (const float*)d_in[6];
    const float* wmk = (const float*)d_in[7];
    const float* wmv = (const float*)d_in[8];
    const float* qnw = (const float*)d_in[9];
    const float* knw = (const float*)d_in[10];

    char* ws = (char*)d_ws;
    size_t off = 0;
    auto carve = [&](size_t bytes) -> void* {
        void* p = ws + off;
        off += (bytes + 255) & ~(size_t)255;
        return p;
    };

    const int QDIM  = NH * HD;    // 2048
    const int KVDIM = NKV * HD;   // 512

    unsigned short* xbf   = (unsigned short*)carve((size_t)S_LEN * HIDN * 2);
    unsigned short* mubf  = (unsigned short*)carve((size_t)S_LEN * HIDN * 2);
    unsigned short* wqbf  = (unsigned short*)carve((size_t)QDIM  * HIDN * 2);
    unsigned short* wmqbf = (unsigned short*)carve((size_t)QDIM  * HIDN * 2);
    unsigned short* wkbf  = (unsigned short*)carve((size_t)KVDIM * HIDN * 2);
    unsigned short* wmkbf = (unsigned short*)carve((size_t)KVDIM * HIDN * 2);
    unsigned short* wvbf  = (unsigned short*)carve((size_t)KVDIM * HIDN * 2);
    unsigned short* wmvbf = (unsigned short*)carve((size_t)KVDIM * HIDN * 2);
    unsigned short* wobf  = (unsigned short*)carve((size_t)HIDN  * QDIM * 2);
    float* qf32 = (float*)carve((size_t)S_LEN * QDIM  * 4);
    float* kf32 = (float*)carve((size_t)S_LEN * KVDIM * 4);
    float* vf32 = (float*)carve((size_t)S_LEN * KVDIM * 4);
    unsigned short* qbf = (unsigned short*)carve((size_t)NH  * S_LEN * HD * 2);
    unsigned short* kbf = (unsigned short*)carve((size_t)NKV * S_LEN * HD * 2);
    unsigned short* vbf = (unsigned short*)carve((size_t)NKV * S_LEN * HD * 2);
    unsigned short* abf = (unsigned short*)carve((size_t)S_LEN * QDIM * 2);

    auto cvt = [&](const float* s, unsigned short* d, int n) {
        cvt_f32_to_bf16<<<dim3((n + 255) / 256), dim3(256), 0, stream>>>(s, d, n);
    };
    cvt(x,   xbf,   S_LEN * HIDN);
    cvt(mu,  mubf,  S_LEN * HIDN);
    cvt(wq,  wqbf,  QDIM  * HIDN);
    cvt(wmq, wmqbf, QDIM  * HIDN);
    cvt(wk,  wkbf,  KVDIM * HIDN);
    cvt(wmk, wmkbf, KVDIM * HIDN);
    cvt(wv,  wvbf,  KVDIM * HIDN);
    cvt(wmv, wmvbf, KVDIM * HIDN);
    cvt(wo,  wobf,  HIDN  * QDIM);

    // fused mu-guided projections: q/k/v = x@W^T + mu@Wm^T
    gemm_nt_bf16<<<dim3(QDIM / BN, S_LEN / BM), dim3(256), 0, stream>>>(
        xbf, wqbf, mubf, wmqbf, qf32, QDIM, HIDN);
    gemm_nt_bf16<<<dim3(KVDIM / BN, S_LEN / BM), dim3(256), 0, stream>>>(
        xbf, wkbf, mubf, wmkbf, kf32, KVDIM, HIDN);
    gemm_nt_bf16<<<dim3(KVDIM / BN, S_LEN / BM), dim3(256), 0, stream>>>(
        xbf, wvbf, mubf, wmvbf, vf32, KVDIM, HIDN);

    // per-head RMSNorm + RoPE + relayout to [H,S,D] bf16 (V: relayout only)
    norm_rope_relayout<<<dim3(S_LEN, NH),  dim3(128), 0, stream>>>(qf32, qnw, qbf, NH, 1);
    norm_rope_relayout<<<dim3(S_LEN, NKV), dim3(128), 0, stream>>>(kf32, knw, kbf, NKV, 1);
    norm_rope_relayout<<<dim3(S_LEN, NKV), dim3(128), 0, stream>>>(vf32, nullptr, vbf, NKV, 0);

    // causal GQA flash attention
    flash_attn_bf16<<<dim3(S_LEN / BR, NH), dim3(128), 0, stream>>>(qbf, kbf, vbf, abf);

    // output projection -> f32 d_out
    gemm_nt_bf16<<<dim3(HIDN / BN, S_LEN / BM), dim3(256), 0, stream>>>(
        abf, wobf, nullptr, nullptr, (float*)d_out, HIDN, HIDN);
}
